// SpectralConv3d_13391708029576
// MI455X (gfx1250) — matrix operand order repaired and checked by reading, not compiled, not run on hardware
//
#include <hip/hip_runtime.h>

// ---------------------------------------------------------------------------
// FNO spectral conv 3D + FFN + LayerNorm for MI455X (gfx1250, wave32, WMMA).
//
// Heavy math via v_wmma_f32_16x16x32_f16 (f32 accum); operands staged in LDS
// as f16 per the CDNA5 WMMA VGPR layouts (CDNA5 ISA 7.12.2).
// Table/weight staging uses GLOBAL_LOAD_ASYNC_TO_LDS_B128 (ASYNCcnt path,
// CDNA5 ISA) + s_wait_asynccnt; streaming input gather
// uses global_prefetch_b8.
// ---------------------------------------------------------------------------

typedef _Float16 h16  __attribute__((ext_vector_type(16)));
typedef float    f32x8 __attribute__((ext_vector_type(8)));
typedef short    s16x8 __attribute__((ext_vector_type(8)));

#define HID 128

// x / S strides (channel-last fp32 tensor [B][M][N][Z][C=32])
#define BSTR  8388608L
#define MSTR  131072L
#define NSTR  2048L
#define ZSTR  32L

// table offsets (in halves) inside the f16 table region of d_ws
#define OFF_CS   0        // stage-A DFT mats  [2][16][64]        (2048)
#define OFF_CI   2048     // stage-C iDFT mat  [64][32]           (2048)
#define OFF_BT   4096     // complex-mix Bt    [3][16][64][64]    (196608)
#define OFF_W0T  200704   // w0 transposed     [128][32]          (4096)
#define OFF_W1T  204800   // w1 transposed     [32][128]          (4096)

#define S_FLOATS 33554432UL              // 4*64^3*32
#define S_BYTES  (S_FLOATS * 4UL)

// LDS row strides (halves) — multiples of 8 for 16B-aligned vector loads
#define XSR 72     // x tile  [512 cols][64 pos]
#define XHS 520    // Xhat    [16 k][512 cols]
#define OSR 40     // O store [512 col2][32 kk]

// ---- async global->LDS 16B copy (per-lane), tracked by ASYNCcnt -----------
#define ASYNC_CP16(ldsPtr, gPtr)                                              \
  asm volatile("global_load_async_to_lds_b128 %0, %1, off"                    \
               :: "v"((unsigned)(size_t)(ldsPtr)),                            \
                  "v"((unsigned long long)(size_t)(gPtr))                     \
               : "memory")
#define ASYNC_WAIT() asm volatile("s_wait_asynccnt 0" ::: "memory")

// ---------------- WMMA fragment helpers (per CDNA5 ISA layouts) -----------

__device__ __forceinline__ f32x8 wmma_f16(h16 a, h16 b, f32x8 c) {
  return __builtin_amdgcn_wmma_f32_16x16x32_f16(false, a, false, b,
                                                (short)0, c, false, false);
}

// A-matrix 16x32 (MxK), row-major source, row stride rs halves, K offset ks*32.
__device__ __forceinline__ h16 ldA(const _Float16* base, int rs, int ks) {
  int lane = threadIdx.x & 31;
  int row  = lane & 15;
  int kb   = (lane >> 4) * 8;
  const _Float16* p = base + row * rs + ks * 32 + kb;
  union { h16 v; s16x8 s[2]; } u;
  u.s[0] = *(const s16x8*)(p);
  u.s[1] = *(const s16x8*)(p + 16);
  return u.v;
}

// B-matrix 32x16 (KxN) from a column-major buffer bt[n][k] (row stride rs).
__device__ __forceinline__ h16 ldB(const _Float16* bt, int rs, int ks) {
  int lane = threadIdx.x & 31;
  int col  = lane & 15;
  int kb   = ((lane >> 4) << 3) + ks * 32;
  const _Float16* p = bt + col * rs + kb;
  union { h16 v; s16x8 s[2]; } u;
  u.s[0] = *(const s16x8*)(p);
  u.s[1] = *(const s16x8*)(p + 16);
  return u.v;
}

// A-matrix 16x32 gathered from an fp32 row-major source (row stride rs floats)
__device__ __forceinline__ h16 ldA_f32(const float* base, int rs) {
  int lane = threadIdx.x & 31;
  int row  = lane & 15;
  int kb   = (lane >> 4) * 8;
  const float* p = base + row * rs;
  h16 a;
#pragma unroll
  for (int j = 0; j < 8; ++j) a[j] = (_Float16)p[kb + j];
#pragma unroll
  for (int j = 0; j < 8; ++j) a[8 + j] = (_Float16)p[16 + kb + j];
  return a;
}

// ---------------- prep: f16 DFT tables, packed complex-mix mats, weights ---

__global__ void fno_prep_kernel(const float* __restrict__ wx,
                                const float* __restrict__ wy,
                                const float* __restrict__ wz,
                                const float* __restrict__ w0,
                                const float* __restrict__ w1,
                                _Float16* __restrict__ tab) {
  const float PI2 = 6.28318530717958647692f;
  int gid = blockIdx.x * blockDim.x + threadIdx.x;
  int nth = gridDim.x * blockDim.x;

  // stage-A forward DFT (ortho): Cf = cos/8, Sf = -sin/8   [2][16][64]
  for (int e = gid; e < 2048; e += nth) {
    int t = e >> 10, k = (e >> 6) & 15, p = e & 63;
    float ang = PI2 * (float)(k * p) / 64.f;
    tab[OFF_CS + e] = (_Float16)(t == 0 ? cosf(ang) * 0.125f
                                        : -sinf(ang) * 0.125f);
  }
  // stage-C inverse DFT [64 pos][32]: k<16 -> Re coeffs, k>=16 -> Im coeffs
  for (int e = gid; e < 2048; e += nth) {
    int p = e >> 5, k = e & 31;
    float v;
    if (k < 16) v = (k == 0) ? 0.125f : 0.25f * cosf(PI2 * (float)(k * p) / 64.f);
    else { int kk = k - 16;
           v = (kk == 0) ? 0.f : -0.25f * sinf(PI2 * (float)(kk * p) / 64.f); }
    tab[OFF_CI + e] = (_Float16)v;
  }
  // packed complex mix, transposed: Bt[ax][mode][o2][kk] = Bk[kk][o2]
  // Bk = [[Wr, Wi], [-Wi, Wr]]   (w layout: [i][o][mode][2])
  for (int e = gid; e < 3 * 16 * 64 * 64; e += nth) {
    int kk = e & 63, o2 = (e >> 6) & 63, k = (e >> 12) & 15, ax = e >> 16;
    const float* w = (ax == 0) ? wx : (ax == 1) ? wy : wz;
    int i = kk & 31, o = o2 & 31;
    int ri; float sgn = 1.f;
    if (kk < 32 && o2 < 32)       { ri = 0; }
    else if (kk >= 32 && o2 < 32) { ri = 1; sgn = -1.f; }
    else if (kk < 32)             { ri = 1; }
    else                          { ri = 0; }
    tab[OFF_BT + e] = (_Float16)(sgn * w[((i * 32 + o) * 16 + k) * 2 + ri]);
  }
  // w0t[h][i]  (w0 is [32][128])
  for (int e = gid; e < 4096; e += nth) {
    int h = e >> 5, i = e & 31;
    tab[OFF_W0T + e] = (_Float16)w0[i * 128 + h];
  }
  // w1t[o][h]  (w1 is [128][32])
  for (int e = gid; e < 4096; e += nth) {
    int o = e >> 7, h = e & 127;
    tab[OFF_W1T + e] = (_Float16)w1[h * 32 + o];
  }
}

// ---------------- generic per-axis spectral kernel -------------------------
__global__ __launch_bounds__(256) void fno_axis_kernel(
    const float* __restrict__ x, float* __restrict__ S,
    const _Float16* __restrict__ tab, const _Float16* __restrict__ btA,
    long astr, long l1str, long l0str, int accumulate) {
  __shared__ __attribute__((aligned(16))) _Float16 xs_s[512 * XSR];
  __shared__ __attribute__((aligned(16))) _Float16 xhr_s[16 * XHS];
  __shared__ __attribute__((aligned(16))) _Float16 xhi_s[16 * XHS];
  __shared__ __attribute__((aligned(16))) _Float16 ost_s[512 * OSR];
  __shared__ __attribute__((aligned(16))) _Float16 csA_s[2 * 16 * 64];
  __shared__ __attribute__((aligned(16))) _Float16 ci2_s[64 * 32];

  const int tid = threadIdx.x, wave = tid >> 5, lane = tid & 31;
  const int b = blockIdx.x >> 8;
  const int tile = blockIdx.x & 255;
  const int lb = tile * 16;                 // flat line base (16 | 64)
  const long lb1 = lb >> 6, lb0 = lb & 63;
  const size_t bbase =
      (size_t)b * BSTR + (size_t)lb1 * l1str + (size_t)lb0 * l0str;

  // ---- async-stage the DFT tables into LDS (8KB, byte-identical copy) ----
  ASYNC_CP16(&csA_s[tid * 8], tab + OFF_CS + tid * 8);
  ASYNC_CP16(&ci2_s[tid * 8], tab + OFF_CI + tid * 8);

  // ---- x tile (fp32 -> f16) into LDS, with look-ahead prefetch ----
  for (int e = tid; e < 16 * 64 * 32; e += 256) {
    int c = e & 31, pos = (e >> 5) & 63, l = e >> 11;
    size_t gi = bbase + (size_t)l * l0str + (size_t)pos * astr + c;
    if (e + 2048 < 16 * 64 * 32) {
      int e2 = e + 2048;                    // same pos/channel, next line (l+1)
      int c2 = e2 & 31, p2 = (e2 >> 5) & 63, l2 = e2 >> 11;
      __builtin_prefetch(&x[bbase + (size_t)l2 * l0str + (size_t)p2 * astr + c2],
                         0, 0);
    }
    xs_s[(l * 32 + c) * XSR + pos] = (_Float16)x[gi];
  }
  ASYNC_WAIT();
  __syncthreads();

  // ---- stage A: Xhat = [Cf;Sf](16x64) * xs(64x512) ----
  for (int q = 0; q < 4; ++q) {
    int nt = wave * 4 + q;
    f32x8 aR = {}; f32x8 aI = {};
#pragma unroll
    for (int ks = 0; ks < 2; ++ks) {
      h16 bm = ldB(xs_s + nt * 16 * XSR, XSR, ks);
      h16 aC = ldA(csA_s, 64, ks);
      h16 aS = ldA(csA_s + 1024, 64, ks);
      aR = wmma_f16(aC, bm, aR);
      aI = wmma_f16(aS, bm, aI);
    }
    int nc = lane & 15, mo = (lane >> 4) * 8;
#pragma unroll
    for (int r = 0; r < 8; ++r) {
      xhr_s[(mo + r) * XHS + nt * 16 + nc] = (_Float16)aR[r];
      xhi_s[(mo + r) * XHS + nt * 16 + nc] = (_Float16)aI[r];
    }
  }
  __syncthreads();

  // ---- stage B: per mode, [ReO|ImO] = [Xr|Xi](16x64) * Bk_t ----
  for (int q = 0; q < 2; ++q) {
    int k = wave * 2 + q;
    for (int nt = 0; nt < 4; ++nt) {
      f32x8 acc = {};
#pragma unroll
      for (int ks = 0; ks < 2; ++ks) {
        const _Float16* xk = (ks == 0 ? xhr_s : xhi_s) + k * XHS; // 16x32 rs=32
        h16 a  = ldA(xk, 32, 0);
        h16 bm = ldB(btA + ((size_t)k * 64 + nt * 16) * 64, 64, ks);
        acc = wmma_f16(a, bm, acc);
      }
      int nc = lane & 15, mo = (lane >> 4) * 8;
      int o = (nt & 1) * 16 + nc;
      int kkC = k + (nt >= 2 ? 16 : 0);
#pragma unroll
      for (int r = 0; r < 8; ++r) {
        int line = mo + r;
        ost_s[(line * 32 + o) * OSR + kkC] = (_Float16)acc[r];
      }
    }
  }
  __syncthreads();

  // ---- stage C: out(64x512) = [CiT|SiT](64x32) * O(32x512) -> S (+)= ----
  for (int q = 0; q < 16; ++q) {
    int f = wave * 16 + q;
    int mt = f >> 5, nt = f & 31;
    f32x8 acc = {};
    h16 a  = ldA(ci2_s + mt * 16 * 32, 32, 0);
    h16 bm = ldB(ost_s + nt * 16 * OSR, OSR, 0);
    acc = wmma_f16(a, bm, acc);
    int nc = lane & 15, mo = (lane >> 4) * 8;
    int col2 = nt * 16 + nc;
    int dl = col2 >> 5, o = col2 & 31;
    size_t gb = bbase + (size_t)dl * l0str + o;
#pragma unroll
    for (int r = 0; r < 8; ++r) {
      size_t gi = gb + (size_t)(mt * 16 + mo + r) * astr;
      if (accumulate) S[gi] += acc[r];
      else            S[gi]  = acc[r];
    }
  }
}

// ---------------- FFN (32->128 relu ->32) + LayerNorm ----------------------
__global__ __launch_bounds__(256) void fno_ffn_kernel(
    const float* __restrict__ S, const _Float16* __restrict__ w0t,
    const _Float16* __restrict__ w1t, const float* __restrict__ b0,
    const float* __restrict__ b1, const float* __restrict__ lng,
    const float* __restrict__ lnb, float* __restrict__ out) {
  __shared__ __attribute__((aligned(16))) _Float16 hs[8][16 * 136];
  __shared__ __attribute__((aligned(16))) float    ys[8][16 * 32];
  __shared__ __attribute__((aligned(16))) _Float16 w0s[HID * 32];
  __shared__ __attribute__((aligned(16))) _Float16 w1s[32 * HID];
  const int tid = threadIdx.x, wave = tid >> 5, lane = tid & 31;
  const int nc = lane & 15, mo = (lane >> 4) * 8;
  const size_t pt0 = ((size_t)blockIdx.x * 8 + wave) * 16;
  const float* srow = S + pt0 * 32;

  // async-stage both weight matrices into LDS (16KB)
  ASYNC_CP16(&w0s[tid * 16],     w0t + tid * 16);
  ASYNC_CP16(&w0s[tid * 16 + 8], w0t + tid * 16 + 8);
  ASYNC_CP16(&w1s[tid * 16],     w1t + tid * 16);
  ASYNC_CP16(&w1s[tid * 16 + 8], w1t + tid * 16 + 8);

  h16 aS = ldA_f32(srow, 32);   // overlaps with async weight staging
  ASYNC_WAIT();
  __syncthreads();

  // layer 0: h = relu(s @ w0 + b0)
  for (int nt = 0; nt < 8; ++nt) {
    float bv = b0[nt * 16 + nc];
    f32x8 acc;
#pragma unroll
    for (int r = 0; r < 8; ++r) acc[r] = bv;
    h16 bm = ldB(w0s + nt * 16 * 32, 32, 0);
    acc = wmma_f16(aS, bm, acc);
#pragma unroll
    for (int r = 0; r < 8; ++r) {
      float v = acc[r]; v = v > 0.f ? v : 0.f;
      hs[wave][(mo + r) * 136 + nt * 16 + nc] = (_Float16)v;
    }
  }
  __syncthreads();

  // layer 1: y = h @ w1 + b1
  for (int nt = 0; nt < 2; ++nt) {
    float bv = b1[nt * 16 + nc];
    f32x8 acc;
#pragma unroll
    for (int r = 0; r < 8; ++r) acc[r] = bv;
#pragma unroll
    for (int ks = 0; ks < 4; ++ks) {
      h16 a  = ldA(hs[wave], 136, ks);
      h16 bm = ldB(w1s + nt * 16 * HID, HID, ks);
      acc = wmma_f16(a, bm, acc);
    }
#pragma unroll
    for (int r = 0; r < 8; ++r)
      ys[wave][(mo + r) * 32 + nt * 16 + nc] = acc[r];
  }
  __syncthreads();

  // LayerNorm over the 32 channels of each point
  if (lane < 16) {
    const float* yr = &ys[wave][lane * 32];
    float s = 0.f, s2 = 0.f;
#pragma unroll
    for (int j = 0; j < 32; ++j) { float v = yr[j]; s += v; s2 += v * v; }
    float mu = s * (1.f / 32.f);
    float var = s2 * (1.f / 32.f) - mu * mu;
    float rstd = rsqrtf(var + 1e-5f);
    float* orow = out + (pt0 + lane) * 32;
#pragma unroll
    for (int j = 0; j < 32; ++j)
      orow[j] = (yr[j] - mu) * rstd * lng[j] + lnb[j];
  }
}

// ---------------------------------------------------------------------------

extern "C" void kernel_launch(void* const* d_in, const int* in_sizes, int n_in,
                              void* d_out, int out_size, void* d_ws,
                              size_t ws_size, hipStream_t stream) {
  (void)in_sizes; (void)n_in; (void)out_size; (void)ws_size;
  const float* x   = (const float*)d_in[0];
  const float* wx  = (const float*)d_in[1];
  const float* wy  = (const float*)d_in[2];
  const float* wz  = (const float*)d_in[3];
  const float* w0  = (const float*)d_in[4];
  const float* b0  = (const float*)d_in[5];
  const float* w1  = (const float*)d_in[6];
  const float* b1  = (const float*)d_in[7];
  const float* lng = (const float*)d_in[8];
  const float* lnb = (const float*)d_in[9];
  float* out = (float*)d_out;

  float*     S   = (float*)d_ws;
  _Float16*  tab = (_Float16*)((char*)d_ws + S_BYTES);

  fno_prep_kernel<<<64, 256, 0, stream>>>(wx, wy, wz, w0, w1, tab);

  const _Float16* bt = tab + OFF_BT;
  // axis M (wx): lines over (n,z)   — writes S
  fno_axis_kernel<<<1024, 256, 0, stream>>>(x, S, tab, bt + 0 * 65536,
                                            MSTR, NSTR, ZSTR, 0);
  // axis N (wy): lines over (m,z)   — accumulates
  fno_axis_kernel<<<1024, 256, 0, stream>>>(x, S, tab, bt + 1 * 65536,
                                            NSTR, MSTR, ZSTR, 1);
  // axis Z (wz): lines over (m,n)   — accumulates
  fno_axis_kernel<<<1024, 256, 0, stream>>>(x, S, tab, bt + 2 * 65536,
                                            ZSTR, MSTR, NSTR, 1);

  fno_ffn_kernel<<<8192, 256, 0, stream>>>(S, tab + OFF_W0T, tab + OFF_W1T,
                                           b0, b1, lng, lnb, out);
}
